// ScaledDotProduct_47742856462955
// MI455X (gfx1250) — compile-verified, matrix operand order checked by reading, not run on hardware
//
#include <hip/hip_runtime.h>
#include <cstddef>
#include <cstdint>

// ---------------------------------------------------------------------------
// MI455X / gfx1250 fused attention:
//   cast/transpose prep -> (X·Wt + b) projections -> Q·K^T -> softmax -> attn·V
// All GEMMs: v_wmma_f32_16x16x32_bf16 (wave32), LDS staged with
// GLOBAL_LOAD_ASYNC_TO_LDS_B128 double buffering (ASYNCcnt).
// Block tile 256x128 (8 waves, 4x2), wave tile 64x64: 16 ds_load_b128 feed
// 16 WMMAs per K-step (1.0 loads/wmma).
// ---------------------------------------------------------------------------

typedef __bf16 bf16;
typedef __attribute__((ext_vector_type(16))) __bf16 v16bf;
typedef __attribute__((ext_vector_type(8)))  __bf16 v8bf;
typedef __attribute__((ext_vector_type(8)))  float  v8f;
typedef __attribute__((ext_vector_type(4)))  float  v4f;
typedef __attribute__((ext_vector_type(4)))  int    v4i;

constexpr int BM = 256;      // block tile M
constexpr int BN = 128;      // block tile N
constexpr int BK = 32;       // K per WMMA step
constexpr int KPAD = BK + 8; // padded LDS row: 40 bf16 = 80B, keeps b128 aligned

#ifdef __has_builtin
#if __has_builtin(__builtin_amdgcn_global_load_async_to_lds_b128)
#define USE_ASYNC_BUILTIN 1
#endif
#if __has_builtin(__builtin_amdgcn_s_wait_asynccnt)
#define USE_WAITASYNC_BUILTIN 1
#endif
#endif

// 16-byte global -> LDS async copy (no VGPR round trip, tracked by ASYNCcnt).
__device__ __forceinline__ void async_cp16(void* lds_generic, const bf16* g) {
  const uint32_t loff = (uint32_t)(uintptr_t)lds_generic;  // low 32 = LDS offset
#ifdef USE_ASYNC_BUILTIN
  __builtin_amdgcn_global_load_async_to_lds_b128(
      (__attribute__((address_space(1))) v4i*)(uintptr_t)g,
      (__attribute__((address_space(3))) v4i*)(uintptr_t)loff,
      0, 0);
#else
  asm volatile("global_load_async_to_lds_b128 %0, %1, off"
               :: "v"(loff), "v"((uint64_t)(uintptr_t)g)
               : "memory");
#endif
}

template <int N>
__device__ __forceinline__ void wait_async() {
#ifdef USE_WAITASYNC_BUILTIN
  __builtin_amdgcn_s_wait_asynccnt(N);
#else
  asm volatile("s_wait_asynccnt %0" :: "i"(N) : "memory");
#endif
}

__device__ __forceinline__ v16bf mk16(v8bf lo, v8bf hi) {
  v16bf r;
#pragma unroll
  for (int i = 0; i < 8; ++i) { r[i] = lo[i]; r[i + 8] = hi[i]; }
  return r;
}

// ---------------------------------------------------------------------------
// Unified batched GEMM, all-bf16 operands:
//   C[b] = scale * (A[b] · B[b]^T) + bias     (B stored [N][K], k contiguous)
// ---------------------------------------------------------------------------
template <typename TC, bool BIAS>
__global__ __launch_bounds__(256) void gemm_bf16_wmma(
    const bf16* __restrict__ A, long long sA, int lda,
    const bf16* __restrict__ Bm, long long sB, int ldb,
    TC* __restrict__ C, long long sC, int ldc,
    const float* __restrict__ bias, float scale, int K) {
  __shared__ __align__(16) bf16 As[2][BM][KPAD];  // 2*256*40*2 = 40 KB
  __shared__ __align__(16) bf16 Bs[2][BN][KPAD];  // 2*128*40*2 = 20 KB

  const int tid = threadIdx.x;
  const int m0 = blockIdx.y * BM;
  const int n0 = blockIdx.x * BN;
  A  += (long long)blockIdx.z * sA;
  Bm += (long long)blockIdx.z * sB;
  C  += (long long)blockIdx.z * sC;

  // Staging map (uniform 6 async b128 per thread per K-step):
  //  A: thread owns full row `tid` (4 chunks of 16B)
  //  B: thread owns half row `tid>>1`, chunk pair (tid&1)
  const int scgB = (tid & 1) << 4;
  const bf16* gA = A  + (size_t)(m0 + tid) * lda;
  const bf16* gB = Bm + (size_t)(n0 + (tid >> 1)) * ldb + scgB;

  const int wid  = tid >> 5;
  const int lane = tid & 31;
  const int half = lane >> 4;        // half-wave selects K sub-range
  const int ln   = lane & 15;        // row (A) / col (B,C) within 16
  const int mW   = (wid >> 1) * 64;  // wave M origin: 0,64,128,192
  const int nW   = (wid & 1) * 64;   // wave N origin: 0,64

  v8f acc[4][4];
#pragma unroll
  for (int i = 0; i < 4; ++i)
#pragma unroll
    for (int j = 0; j < 4; ++j)
#pragma unroll
      for (int e = 0; e < 8; ++e) acc[i][j][e] = 0.0f;

  auto stage = [&](int buf, int k0) {
    const bf16* ga = gA + k0;
    bf16* la = &As[buf][tid][0];
    async_cp16(la,      ga);
    async_cp16(la + 8,  ga + 8);
    async_cp16(la + 16, ga + 16);
    async_cp16(la + 24, ga + 24);
    const bf16* gb = gB + k0;
    bf16* lb = &Bs[buf][tid >> 1][scgB];
    async_cp16(lb,     gb);
    async_cp16(lb + 8, gb + 8);
  };

  auto compute = [&](int cur) {
    // Fragment layout per ISA 7.12.2 (16-bit A 16x32):
    // lane ln(+16*half): elems 0..7 -> k = half*8+e, elems 8..15 -> k = 16+half*8+e.
    v16bf afr[4], bfr[4];
#pragma unroll
    for (int mf = 0; mf < 4; ++mf) {
      const bf16* p = &As[cur][mW + mf * 16 + ln][half * 8];
      afr[mf] = mk16(*(const v8bf*)p, *(const v8bf*)(p + 16));
    }
#pragma unroll
    for (int nf = 0; nf < 4; ++nf) {
      const bf16* p = &Bs[cur][nW + nf * 16 + ln][half * 8];
      bfr[nf] = mk16(*(const v8bf*)p, *(const v8bf*)(p + 16));
    }
#pragma unroll
    for (int mf = 0; mf < 4; ++mf)
#pragma unroll
      for (int nf = 0; nf < 4; ++nf)
        acc[mf][nf] = __builtin_amdgcn_wmma_f32_16x16x32_bf16(
            false, afr[mf], false, bfr[nf], (short)0, acc[mf][nf], false, false);
  };

  // Double-buffered main loop, last iteration peeled (branch-free steady state).
  stage(0, 0);
  int cur = 0;
  int k0 = 0;
  for (; k0 + BK < K; k0 += BK, cur ^= 1) {
    stage(cur ^ 1, k0 + BK);   // 6 more in flight
    wait_async<6>();           // current buffer landed (FIFO)
    __syncthreads();
    compute(cur);
    __syncthreads();
  }
  wait_async<0>();
  __syncthreads();
  compute(cur);

  // Epilogue. C layout: VGPR e -> m = half*8 + e, n = ln.
#pragma unroll
  for (int nf = 0; nf < 4; ++nf) {
    const int col = n0 + nW + nf * 16 + ln;
    float bv = 0.0f;
    if constexpr (BIAS) bv = bias[col];
#pragma unroll
    for (int mf = 0; mf < 4; ++mf) {
#pragma unroll
      for (int e = 0; e < 8; ++e) {
        const int rowg = m0 + mW + mf * 16 + half * 8 + e;
        C[(size_t)rowg * ldc + col] = (TC)(acc[mf][nf][e] * scale + bv);
      }
    }
  }
}

// f32 -> bf16 elementwise cast (8 elems/thread).
__global__ __launch_bounds__(256) void cast_f32_bf16(
    const float* __restrict__ in, bf16* __restrict__ out) {
  const size_t i = ((size_t)blockIdx.x * 256 + threadIdx.x) * 8;
  const v4f a = ((const v4f*)(in + i))[0];
  const v4f b = ((const v4f*)(in + i))[1];
  v8bf o;
#pragma unroll
  for (int j = 0; j < 4; ++j) { o[j] = (bf16)a[j]; o[j + 4] = (bf16)b[j]; }
  *(v8bf*)(out + i) = o;
}

// Transpose [R x C] (f32 or bf16) -> bf16 [C x R], 64x64 LDS tiles.
template <typename T>
__global__ __launch_bounds__(256) void transpose_to_bf16(
    const T* __restrict__ in, bf16* __restrict__ out,
    int R, int Cc, long long sIn, long long sOut) {
  __shared__ __align__(16) bf16 t[64][72];
  const int tid = threadIdx.x;
  in  += (long long)blockIdx.z * sIn;
  out += (long long)blockIdx.z * sOut;
  const int r0 = blockIdx.y * 64;
  const int c0 = blockIdx.x * 64;
  const int row = tid >> 2;
  const int cg  = (tid & 3) << 4;
  const T* s = in + (size_t)(r0 + row) * Cc + c0 + cg;
  if constexpr (sizeof(T) == 4) {
    const v4f* sv = (const v4f*)s;
    v4f a = sv[0], b = sv[1], c = sv[2], d = sv[3];
    v8bf lo, hi;
#pragma unroll
    for (int j = 0; j < 4; ++j) {
      lo[j] = (bf16)a[j]; lo[j + 4] = (bf16)b[j];
      hi[j] = (bf16)c[j]; hi[j + 4] = (bf16)d[j];
    }
    *(v8bf*)&t[row][cg] = lo;
    *(v8bf*)&t[row][cg + 8] = hi;
  } else {
    const v8bf* sv = (const v8bf*)s;
    *(v8bf*)&t[row][cg] = sv[0];
    *(v8bf*)&t[row][cg + 8] = sv[1];
  }
  __syncthreads();
  v8bf o0, o1;
#pragma unroll
  for (int j = 0; j < 8; ++j) {
    o0[j] = t[cg + j][row];
    o1[j] = t[cg + 8 + j][row];
  }
  bf16* dp = out + (size_t)(c0 + row) * R + r0 + cg;
  *(v8bf*)dp = o0;
  *(v8bf*)(dp + 8) = o1;
}

// Row softmax over S logits (f32) + mask, emit bf16 probabilities.
__global__ __launch_bounds__(256) void softmax_bf16_kernel(
    const float* __restrict__ logits, const float* __restrict__ mask,
    bf16* __restrict__ probs, int S) {
  __shared__ float red[256];
  const int tid = threadIdx.x;
  const size_t row = blockIdx.x;
  const float* l = logits + row * S;
  const float* m = mask + row * S;
  bf16* p = probs + row * S;

  float x[8];
  float mx = -3.0e38f;
#pragma unroll
  for (int i = 0; i < 8; ++i) {
    const int idx = i * 256 + tid;
    x[i] = l[idx] + m[idx];
    mx = fmaxf(mx, x[i]);
  }
  red[tid] = mx;
  __syncthreads();
  for (int s = 128; s > 0; s >>= 1) {
    if (tid < s) red[tid] = fmaxf(red[tid], red[tid + s]);
    __syncthreads();
  }
  mx = red[0];
  __syncthreads();

  float sum = 0.0f;
#pragma unroll
  for (int i = 0; i < 8; ++i) { x[i] = __expf(x[i] - mx); sum += x[i]; }
  red[tid] = sum;
  __syncthreads();
  for (int s = 128; s > 0; s >>= 1) {
    if (tid < s) red[tid] += red[tid + s];
    __syncthreads();
  }
  const float inv = 1.0f / red[0];
#pragma unroll
  for (int i = 0; i < 8; ++i) p[i * 256 + tid] = (bf16)(x[i] * inv);
}

extern "C" void kernel_launch(void* const* d_in, const int* in_sizes, int n_in,
                              void* d_out, int out_size, void* d_ws, size_t ws_size,
                              hipStream_t stream) {
  (void)in_sizes; (void)n_in; (void)out_size; (void)ws_size;

  const float* q    = (const float*)d_in[0];
  const float* k    = (const float*)d_in[1];
  const float* v    = (const float*)d_in[2];
  const float* mask = (const float*)d_in[3];
  const float* Wq   = (const float*)d_in[4];
  const float* bq   = (const float*)d_in[5];
  const float* Wk   = (const float*)d_in[6];
  const float* bk   = (const float*)d_in[7];
  const float* Wv   = (const float*)d_in[8];
  const float* bv   = (const float*)d_in[9];
  float* out = (float*)d_out;

  constexpr int Bn = 4, S = 2048, D = 1024;
  constexpr size_t MB = (size_t)1 << 20;

  // Workspace layout (216 MB):
  char* ws = (char*)d_ws;
  bf16* qb  = (bf16*)(ws);             // 16 MB  bf16 q
  bf16* kb  = (bf16*)(ws + 16 * MB);   // 16 MB
  bf16* vb  = (bf16*)(ws + 32 * MB);   // 16 MB
  bf16* WqT = (bf16*)(ws + 48 * MB);   //  2 MB  Wq^T  [D_model][D_in]
  bf16* WkT = (bf16*)(ws + 50 * MB);   //  2 MB
  bf16* WvT = (bf16*)(ws + 52 * MB);   //  2 MB
  bf16* Qp  = (bf16*)(ws + 56 * MB);   // 16 MB  projected Q
  bf16* Kp  = (bf16*)(ws + 72 * MB);   // 16 MB
  bf16* Vp  = (bf16*)(ws + 88 * MB);   // 16 MB
  bf16* VpT = (bf16*)(ws + 104 * MB);  // 16 MB  Vp^T [D][S] per batch
  float* lg = (float*)(ws + 120 * MB); // 64 MB  logits
  bf16* pr  = (bf16*)(ws + 184 * MB);  // 32 MB  probabilities

  const dim3 blk(256);

  // 0) f32 -> bf16 casts of activations; cast+transpose of weights.
  const dim3 gc((Bn * S * D) / (256 * 8));
  cast_f32_bf16<<<gc, blk, 0, stream>>>(q, qb);
  cast_f32_bf16<<<gc, blk, 0, stream>>>(k, kb);
  cast_f32_bf16<<<gc, blk, 0, stream>>>(v, vb);
  const dim3 gw(D / 64, D / 64, 1);
  transpose_to_bf16<float><<<gw, blk, 0, stream>>>(Wq, WqT, D, D, 0, 0);
  transpose_to_bf16<float><<<gw, blk, 0, stream>>>(Wk, WkT, D, D, 0, 0);
  transpose_to_bf16<float><<<gw, blk, 0, stream>>>(Wv, WvT, D, D, 0, 0);

  // 1) Projections: [B*S x D] = [B*S x D] · W^T[N][K] + bias, bf16 out.
  const dim3 gp(D / BN, (Bn * S) / BM, 1);
  gemm_bf16_wmma<bf16, true><<<gp, blk, 0, stream>>>(
      qb, 0, D, WqT, 0, D, Qp, 0, D, bq, 1.0f, D);
  gemm_bf16_wmma<bf16, true><<<gp, blk, 0, stream>>>(
      kb, 0, D, WkT, 0, D, Kp, 0, D, bk, 1.0f, D);
  gemm_bf16_wmma<bf16, true><<<gp, blk, 0, stream>>>(
      vb, 0, D, WvT, 0, D, Vp, 0, D, bv, 1.0f, D);

  // 1b) Vp -> Vp^T per batch for the attn·V GEMM.
  const dim3 gt(D / 64, S / 64, Bn);
  transpose_to_bf16<bf16><<<gt, blk, 0, stream>>>(
      Vp, VpT, S, D, (long long)S * D, (long long)S * D);

  // 2) logits[b] = (Qp[b] · Kp[b]^T) / 32.
  const dim3 gl(S / BN, S / BM, Bn);
  gemm_bf16_wmma<float, false><<<gl, blk, 0, stream>>>(
      Qp, (long long)S * D, D, Kp, (long long)S * D, D,
      lg, (long long)S * S, S, nullptr, 0.03125f, D);

  // 3) softmax rows (+mask), bf16 probabilities.
  softmax_bf16_kernel<<<dim3(Bn * S), blk, 0, stream>>>(lg, mask, pr, S);

  // 4) out[b] = probs[b] · (Vp^T[b])^T, K = S, f32 output.
  const dim3 go(D / BN, S / BM, Bn);
  gemm_bf16_wmma<float, false><<<go, blk, 0, stream>>>(
      pr, (long long)S * S, S, VpT, (long long)S * D, S,
      out, (long long)S * D, D, nullptr, 1.0f, S);
}
